// FJMPAttentionTrajectoryDecoder_45861660787501
// MI455X (gfx1250) — matrix operand order repaired and checked by reading, not compiled, not run on hardware
//
#include <hip/hip_runtime.h>
#include <hip/hip_bf16.h>
#include <math.h>

typedef __bf16 bf16;
typedef __attribute__((ext_vector_type(16))) __bf16 v16bf;
typedef __attribute__((ext_vector_type(8)))  __bf16 v8bf;
typedef __attribute__((ext_vector_type(8)))  float  v8f;

#define N_NODES 16000
#define N_EDGES 64000
#define NHEAD   2
#define MODES   6
#define TSTEP   30
#define HD      128
#define KM      (NHEAD*MODES)      /* 12 */
#define EPSGN   1e-5f

/* ---------------- input index map (setup_inputs dict insertion order, params recursed) ---- */
enum {
  IN_VN=0, IN_FD, IN_AT, IN_CTRS, IN_ROT, IN_ORIG, IN_SRC, IN_DST,
  IN_ATW, IN_ATB,
  IN_E_L1W, IN_E_L1B, IN_E_N1W, IN_E_N1B, IN_E_L2W, IN_E_L2B, IN_E_N2W, IN_E_N2B,
  IN_E_TW, IN_E_TB, IN_E_TNW, IN_E_TNB,
  IN_FC1, IN_FC2, IN_FC3,
  IN_WIH, IN_WHH, IN_BIH, IN_BHH,
  IN_F_L1W, IN_F_L1B, IN_F_N1W, IN_F_N1B, IN_F_L2W, IN_F_L2B, IN_F_N2W, IN_F_N2B,
  IN_F_TW, IN_F_TB, IN_F_TNW, IN_F_TNB, IN_F_OW, IN_F_OB
};

/* ---------------- device helpers ---------------- */
__device__ inline float eluf(float x){ return x > 0.f ? x : __expf(x) - 1.f; }

__device__ inline unsigned fkey(float f){
  unsigned u = __float_as_uint(f);
  return (u & 0x80000000u) ? ~u : (u | 0x80000000u);
}
__device__ inline float funkey(unsigned k){
  unsigned u = (k & 0x80000000u) ? (k ^ 0x80000000u) : ~k;
  return __uint_as_float(u);
}

__device__ inline v8f wmma_bf16(v16bf a, v16bf b, v8f c){
  return __builtin_amdgcn_wmma_f32_16x16x32_bf16(false, a, false, b, (short)0, c, false, false);
}

/* A fragment: 16x32 bf16 from row-major LDS. Per ISA 7.12.2, lane (hi,r) holds two
   contiguous K-runs of 8: [kb..kb+7] and [kb+16..kb+23] -> two 16B vector loads. */
__device__ inline v16bf load_a_frag(const bf16* a_rm, int ldm, int k0, int lane){
  int r  = lane & 15;
  int kb = k0 + ((lane & 16) ? 8 : 0);
  const bf16* p = a_rm + r*ldm + kb;
  v8bf lo = *(const v8bf*)(p);
  v8bf hi = *(const v8bf*)(p + 16);
  return __builtin_shufflevector(lo, hi, 0,1,2,3,4,5,6,7,8,9,10,11,12,13,14,15);
}

/* B panels are pre-swizzled fragment-major: for tile (nt,kt), lane holds 16 contiguous
   bf16 (column n = nt*16+(lane&15), k = kt*32 + ((lane&16)>>1) + {0..7, 16..23}). One 32B load. */
__device__ inline size_t bsw_idx(int k, int n, int Kpad){
  int nt = n >> 4, kt = k >> 5, kw = k & 31;
  int lane = (((kw >> 3) & 1) << 4) | (n & 15);
  int e = (kw & 7) | ((kw & 16) >> 1);
  return ((size_t)(nt*(Kpad >> 5) + kt)*32 + lane)*16 + e;
}
__device__ inline v16bf load_b_frag(const bf16* B, int Kpad, int n0, int k0, int lane){
  const bf16* p = B + ((size_t)((n0 >> 4)*(Kpad >> 5) + (k0 >> 5))*32 + lane)*16;
  return *(const v16bf*)p;
}

/* C fragment store, row-major f32 (ISA 7.12.2 32-bit C/D layout) */
__device__ inline void store_c_frag(float* dst, int ldc, int col0, const v8f& c, int lane){
  int col = col0 + (lane & 15);
  int rb  = (lane & 16) ? 8 : 0;
#pragma unroll
  for (int r = 0; r < 8; ++r) dst[(rb + r)*ldc + col] = c[r];
}

__device__ inline void gn_norm4(float v0, float v1, float v2, float v3, float* o){
  float mu = 0.25f*(v0+v1+v2+v3);
  float d0 = v0-mu, d1 = v1-mu, d2 = v2-mu, d3 = v3-mu;
  float var = 0.25f*(d0*d0 + d1*d1 + d2*d2 + d3*d3);
  float rs = rsqrtf(var + EPSGN);
  o[0]=d0*rs; o[1]=d1*rs; o[2]=d2*rs; o[3]=d3*rs;
}

/* ============== prep kernels ============== */

/* W[Nout][srcK] row-major f32 -> fragment-major bf16 panel (Kpad x NoutPad tiles) */
__global__ void k_prep_bmat(const float* __restrict__ W, bf16* __restrict__ dst,
                            int Nout, int srcK, int Kuse, int NoutPad, int Kpad){
  int i = blockIdx.x*blockDim.x + threadIdx.x;
  if (i >= Kpad*NoutPad) return;
  int k = i / NoutPad, o = i % NoutPad;
  float v = (k < Kuse && o < Nout) ? W[o*srcK + k] : 0.f;
  dst[bsw_idx(k, o, Kpad)] = (bf16)v;
}

/* u1[k,h] = sum_o fc3[k,o]*fc1[k,o,h] ; u2[k,h] = sum_o fc3[k,HD+o]*fc2[k,o,h] */
__global__ void k_prep_u(const float* __restrict__ fc1, const float* __restrict__ fc2,
                         const float* __restrict__ fc3, float* __restrict__ u1, float* __restrict__ u2){
  int i = blockIdx.x*blockDim.x + threadIdx.x;
  if (i >= NHEAD*HD) return;
  int k = i / HD, h = i % HD;
  float s1 = 0.f, s2 = 0.f;
  for (int o = 0; o < HD; ++o){
    s1 += fc3[k*2*HD + o]      * fc1[(k*HD + o)*HD + h];
    s2 += fc3[k*2*HD + HD + o] * fc2[(k*HD + o)*HD + h];
  }
  u1[i] = s1; u2[i] = s2;
}

/* Bfc fragment-major panel (K=256, N=128): B[(k*HD+h)][o] = fc1[k][o][h] */
__global__ void k_prep_bfc(const float* __restrict__ fc1, bf16* __restrict__ dst){
  int i = blockIdx.x*blockDim.x + threadIdx.x;
  if (i >= NHEAD*HD*HD) return;
  int k = i/(HD*HD), r = i%(HD*HD), o = r/HD, h = r%HD;
  dst[bsw_idx(k*HD + h, o, 2*HD)] = (bf16)fc1[(k*HD + o)*HD + h];
}

/* mode bias: mb[m*HD+o] = W[o*(HD+MODES) + HD + m]  (one-hot eye(M) folded into bias) */
__global__ void k_prep_modebias(const float* __restrict__ W, float* __restrict__ mb){
  int i = blockIdx.x*blockDim.x + threadIdx.x;
  if (i >= MODES*HD) return;
  int m = i/HD, o = i%HD;
  mb[i] = W[o*(HD+MODES) + HD + m];
}

__global__ void k_f32_to_bf16(const float* __restrict__ x, bf16* __restrict__ y, long n){
  long i = (long)blockIdx.x*blockDim.x + threadIdx.x;
  if (i < n) y[i] = (bf16)x[i];
}

/* ============== edge encoder: pred_rel -> LinearRes -> +agent-type ============== */
struct EncP {
  const float *fd, *ctrs, *at, *atW, *atB;
  const int *src, *dst;
  const bf16 *l1W, *l2W, *tW;
  const float *l1b, *n1w, *n1b, *l2b, *n2w, *n2b, *tb, *tnw, *tnb;
  bf16 *feats;
};

__global__ __launch_bounds__(256) void k_edge_enc(EncP P){
  __shared__ __align__(16) bf16  sX[16][64];
  __shared__ __align__(16) float sAct[16][HD];
  __shared__ __align__(16) bf16  sActB[16][HD];
  __shared__ __align__(16) float sN2[16][HD];

  int tid = threadIdx.x, lane = tid & 31, w = tid >> 5;
  long row0 = (long)blockIdx.x * 16;

  /* stage x = fd[src] - ctrs[dst], 60 cols padded to 64, bf16 */
  for (int i = tid; i < 16*64; i += 256){
    int r = i >> 6, k = i & 63;
    long R = row0 + r;
    int e = (int)(R / MODES), m = (int)(R % MODES);
    float v = 0.f;
    if (k < 2*TSTEP){
      int t = k >> 1, c = k & 1;
      v = P.fd[(((long)P.src[e]*MODES + m)*TSTEP + t)*2 + c] - P.ctrs[P.dst[e]*2 + c];
    }
    sX[r][k] = (bf16)v;
  }
  __syncthreads();

  int col0 = w * 16;

  /* l1: [16x64] @ [64x128] */
  v8f acc = (v8f)0.0f;
  for (int k0 = 0; k0 < 64; k0 += 32){
    v16bf a = load_a_frag(&sX[0][0], 64, k0, lane);
    v16bf b = load_b_frag(P.l1W, 64, col0, k0, lane);
    acc = wmma_bf16(a, b, acc);
  }
  store_c_frag(&sAct[0][0], HD, col0, acc, lane);
  __syncthreads();

  /* GN1 (+bias) + ELU -> bf16 */
  for (int i = tid; i < 16*32; i += 256){
    int r = i >> 5, g = i & 31, c0 = g*4;
    float nd[4];
    gn_norm4(sAct[r][c0+0]+P.l1b[c0+0], sAct[r][c0+1]+P.l1b[c0+1],
             sAct[r][c0+2]+P.l1b[c0+2], sAct[r][c0+3]+P.l1b[c0+3], nd);
#pragma unroll
    for (int j = 0; j < 4; ++j){
      int c = c0 + j;
      sActB[r][c] = (bf16)eluf(nd[j]*P.n1w[c] + P.n1b[c]);
    }
  }
  __syncthreads();

  /* l2: [16x128] @ [128x128]  |  t: [16x64] @ [64x128] */
  v8f acc2 = (v8f)0.0f;
  for (int k0 = 0; k0 < HD; k0 += 32){
    v16bf a = load_a_frag(&sActB[0][0], HD, k0, lane);
    v16bf b = load_b_frag(P.l2W, HD, col0, k0, lane);
    acc2 = wmma_bf16(a, b, acc2);
  }
  v8f acct = (v8f)0.0f;
  for (int k0 = 0; k0 < 64; k0 += 32){
    v16bf a = load_a_frag(&sX[0][0], 64, k0, lane);
    v16bf b = load_b_frag(P.tW, 64, col0, k0, lane);
    acct = wmma_bf16(a, b, acct);
  }

  store_c_frag(&sAct[0][0], HD, col0, acc2, lane);
  __syncthreads();
  /* GN2 (no ELU) -> sN2 */
  for (int i = tid; i < 16*32; i += 256){
    int r = i >> 5, g = i & 31, c0 = g*4;
    float nd[4];
    gn_norm4(sAct[r][c0+0]+P.l2b[c0+0], sAct[r][c0+1]+P.l2b[c0+1],
             sAct[r][c0+2]+P.l2b[c0+2], sAct[r][c0+3]+P.l2b[c0+3], nd);
#pragma unroll
    for (int j = 0; j < 4; ++j){
      int c = c0 + j;
      sN2[r][c] = nd[j]*P.n2w[c] + P.n2b[c];
    }
  }
  __syncthreads();

  store_c_frag(&sAct[0][0], HD, col0, acct, lane);
  __syncthreads();
  /* GN(t) -> elu(n2+nt) + agent-type linear -> feats (bf16) */
  for (int i = tid; i < 16*32; i += 256){
    int r = i >> 5, g = i & 31, c0 = g*4;
    long R = row0 + r;
    int e = (int)(R / MODES);
    int se = P.src[e], de = P.dst[e];
    float a0 = P.at[se*2+0], a1 = P.at[se*2+1], a2 = P.at[de*2+0], a3 = P.at[de*2+1];
    float nd[4];
    gn_norm4(sAct[r][c0+0]+P.tb[c0+0], sAct[r][c0+1]+P.tb[c0+1],
             sAct[r][c0+2]+P.tb[c0+2], sAct[r][c0+3]+P.tb[c0+3], nd);
#pragma unroll
    for (int j = 0; j < 4; ++j){
      int c = c0 + j;
      float nt  = nd[j]*P.tnw[c] + P.tnb[c];
      float val = eluf(sN2[r][c] + nt);
      float atv = a0*P.atW[c*4+0] + a1*P.atW[c*4+1] + a2*P.atW[c*4+2] + a3*P.atW[c*4+3] + P.atB[c];
      P.feats[R*HD + c] = (bf16)(val + atv);
    }
  }
}

/* ============== attention scores: e = leaky(u1.feats + u2.v_n[dst]), + segment max ======= */
__global__ void k_escore(const bf16* __restrict__ feats, const float* __restrict__ vN,
                         const int* __restrict__ dst, const float* __restrict__ u1,
                         const float* __restrict__ u2, float* __restrict__ eo,
                         unsigned* __restrict__ mxkey){
  int i = blockIdx.x*blockDim.x + threadIdx.x;
  if (i >= N_EDGES*MODES) return;
  int e = i / MODES, m = i % MODES;
  int d = dst[e];
  const bf16*  f  = feats + (long)i*HD;
  const float* vn = vN + ((long)d*MODES + m)*HD;
  float a0=0.f, a1=0.f, b0=0.f, b1=0.f;
  for (int h0 = 0; h0 < HD; h0 += 8){
    v8bf f8 = *(const v8bf*)(f + h0);
    float4 v0 = *(const float4*)(vn + h0);
    float4 v1 = *(const float4*)(vn + h0 + 4);
#pragma unroll
    for (int j = 0; j < 8; ++j){
      float fv = (float)f8[j];
      a0 += u1[h0+j]*fv;  a1 += u1[HD+h0+j]*fv;
    }
    b0 += u2[h0+0]*v0.x + u2[h0+1]*v0.y + u2[h0+2]*v0.z + u2[h0+3]*v0.w
        + u2[h0+4]*v1.x + u2[h0+5]*v1.y + u2[h0+6]*v1.z + u2[h0+7]*v1.w;
    b1 += u2[HD+h0+0]*v0.x + u2[HD+h0+1]*v0.y + u2[HD+h0+2]*v0.z + u2[HD+h0+3]*v0.w
        + u2[HD+h0+4]*v1.x + u2[HD+h0+5]*v1.y + u2[HD+h0+6]*v1.z + u2[HD+h0+7]*v1.w;
  }
  float e0 = a0 + b0, e1 = a1 + b1;
  e0 = e0 > 0.f ? e0 : 0.2f*e0;
  e1 = e1 > 0.f ? e1 : 0.2f*e1;
  eo[(e*NHEAD + 0)*MODES + m] = e0;
  eo[(e*NHEAD + 1)*MODES + m] = e1;
  atomicMax(&mxkey[(d*NHEAD + 0)*MODES + m], fkey(e0));
  atomicMax(&mxkey[(d*NHEAD + 1)*MODES + m], fkey(e1));
}

__global__ void k_expsum(const int* __restrict__ dst, const unsigned* __restrict__ mxkey,
                         float* __restrict__ eio, float* __restrict__ den){
  int i = blockIdx.x*blockDim.x + threadIdx.x;
  if (i >= N_EDGES*KM) return;
  int e = i / KM, km = i % KM;
  int d = dst[e];
  float ee = __expf(eio[i] - funkey(mxkey[d*KM + km]));
  eio[i] = ee;
  atomicAdd(&den[d*KM + km], ee);
}

/* s[dst,k,m,:] += 0.5 * alpha * feats[e,m,:]   (0.5 folds the head-mean) */
__global__ __launch_bounds__(128) void k_scatter(const bf16* __restrict__ feats,
                                                 const float* __restrict__ ee,
                                                 const float* __restrict__ den,
                                                 const int* __restrict__ dst,
                                                 float* __restrict__ s){
  int pair = blockIdx.x;                 /* (e,m) */
  int h = threadIdx.x;
  int e = pair / MODES, m = pair % MODES;
  int d = dst[e];
  __shared__ float coef[NHEAD];
  if (h < NHEAD)
    coef[h] = 0.5f * ee[e*KM + h*MODES + m] / den[d*KM + h*MODES + m];
  __syncthreads();
  float fv = (float)feats[(long)pair*HD + h];
  atomicAdd(&s[(((long)d*NHEAD + 0)*MODES + m)*HD + h], coef[0]*fv);
  atomicAdd(&s[(((long)d*NHEAD + 1)*MODES + m)*HD + h], coef[1]*fv);
}

/* pack s -> bf16 rows [(n,m)][head0(128) | head1(128)] */
__global__ void k_pack_s(const float* __restrict__ s, bf16* __restrict__ sb){
  long i = (long)blockIdx.x*blockDim.x + threadIdx.x;
  if (i >= (long)N_NODES*MODES*2*HD) return;
  int  kk  = (int)(i % (2*HD));
  long row = i / (2*HD);
  int n = (int)(row / MODES), m = (int)(row % MODES);
  int head = kk / HD, h = kk % HD;
  sb[i] = (bf16)s[(((long)n*NHEAD + head)*MODES + m)*HD + h];
}

/* ============== generic WMMA GEMM: C[rows,Nout] = A_bf16[rows,K] @ B_bf16[K,Nout] + bias == */
__global__ __launch_bounds__(256) void k_gemm_bf16(const bf16* __restrict__ A, int K,
                                                   const bf16* __restrict__ B,
                                                   const float* __restrict__ bias,
                                                   float* Cf, bf16* Cb, int Nout){
  extern __shared__ __align__(16) bf16 sA[];   /* 16 x K */
  int tid = threadIdx.x, lane = tid & 31, w = tid >> 5;
  long row0 = (long)blockIdx.x * 16;
  int col0 = blockIdx.y*128 + w*16;
  for (int i = tid; i < 16*K; i += 256){
    int r = i / K, k = i % K;
    sA[i] = A[(row0 + r)*(long)K + k];
  }
  __syncthreads();
  v8f acc = (v8f)0.0f;
  for (int k0 = 0; k0 < K; k0 += 32){
    v16bf a = load_a_frag(sA, K, k0, lane);
    v16bf b = load_b_frag(B, K, col0, k0, lane);
    acc = wmma_bf16(a, b, acc);
  }
  int col = col0 + (lane & 15);
  int rb  = (lane & 16) ? 8 : 0;
  float bv = bias ? bias[col] : 0.f;
#pragma unroll
  for (int r = 0; r < 8; ++r){
    float v = acc[r] + bv;
    long idx = (row0 + rb + r)*(long)Nout + col;
    if (Cf) Cf[idx] = v;
    if (Cb) Cb[idx] = (bf16)v;
  }
}

/* ============== GRU gates + has_in select ============== */
__global__ void k_gru(const float* __restrict__ gi, const float* __restrict__ gh,
                      const float* __restrict__ vN, const float* __restrict__ den,
                      bf16* __restrict__ vnoutB){
  long i = (long)blockIdx.x*blockDim.x + threadIdx.x;
  if (i >= (long)N_NODES*MODES*HD) return;
  long row = i / HD; int h = (int)(i % HD);
  int n = (int)(row / MODES);
  float ir = gi[row*3*HD + h], iz = gi[row*3*HD + HD + h], in = gi[row*3*HD + 2*HD + h];
  float hr = gh[row*3*HD + h], hz = gh[row*3*HD + HD + h], hn = gh[row*3*HD + 2*HD + h];
  float r = 1.f/(1.f + __expf(-(ir + hr)));
  float z = 1.f/(1.f + __expf(-(iz + hz)));
  float nn = tanhf(in + r*hn);
  float hprev = vN[i];
  float hnew = (1.f - z)*nn + z*hprev;
  bool has_in = den[(long)n*KM] > 0.f;
  vnoutB[i] = (bf16)(has_in ? hnew : hprev);
}

/* ============== fout LinearRes (eye(M) folded into per-mode bias) ============== */
struct FoutP {
  const bf16 *X;
  const bf16 *l1W, *l2W, *tW;
  const float *l1b, *l1mb, *n1w, *n1b, *l2b, *n2w, *n2b, *tb, *tmb, *tnw, *tnb;
  bf16 *out;
};

__global__ __launch_bounds__(256) void k_fout_enc(FoutP P){
  __shared__ __align__(16) bf16  sX[16][HD];
  __shared__ __align__(16) float sAct[16][HD];
  __shared__ __align__(16) bf16  sActB[16][HD];
  __shared__ __align__(16) float sN2[16][HD];
  int tid = threadIdx.x, lane = tid & 31, w = tid >> 5;
  long row0 = (long)blockIdx.x * 16;

  for (int i = tid; i < 16*HD; i += 256){
    int r = i >> 7, k = i & 127;
    sX[r][k] = P.X[(row0 + r)*HD + k];
  }
  __syncthreads();

  int col0 = w * 16;
  v8f acc = (v8f)0.0f;
  for (int k0 = 0; k0 < HD; k0 += 32){
    v16bf a = load_a_frag(&sX[0][0], HD, k0, lane);
    v16bf b = load_b_frag(P.l1W, HD, col0, k0, lane);
    acc = wmma_bf16(a, b, acc);
  }
  store_c_frag(&sAct[0][0], HD, col0, acc, lane);
  __syncthreads();

  for (int i = tid; i < 16*32; i += 256){
    int r = i >> 5, g = i & 31, c0 = g*4;
    int m = (int)((row0 + r) % MODES);
    const float* mb = P.l1mb + m*HD;
    float nd[4];
    gn_norm4(sAct[r][c0+0]+P.l1b[c0+0]+mb[c0+0], sAct[r][c0+1]+P.l1b[c0+1]+mb[c0+1],
             sAct[r][c0+2]+P.l1b[c0+2]+mb[c0+2], sAct[r][c0+3]+P.l1b[c0+3]+mb[c0+3], nd);
#pragma unroll
    for (int j = 0; j < 4; ++j){
      int c = c0 + j;
      sActB[r][c] = (bf16)eluf(nd[j]*P.n1w[c] + P.n1b[c]);
    }
  }
  __syncthreads();

  v8f acc2 = (v8f)0.0f;
  for (int k0 = 0; k0 < HD; k0 += 32){
    v16bf a = load_a_frag(&sActB[0][0], HD, k0, lane);
    v16bf b = load_b_frag(P.l2W, HD, col0, k0, lane);
    acc2 = wmma_bf16(a, b, acc2);
  }
  v8f acct = (v8f)0.0f;
  for (int k0 = 0; k0 < HD; k0 += 32){
    v16bf a = load_a_frag(&sX[0][0], HD, k0, lane);
    v16bf b = load_b_frag(P.tW, HD, col0, k0, lane);
    acct = wmma_bf16(a, b, acct);
  }

  store_c_frag(&sAct[0][0], HD, col0, acc2, lane);
  __syncthreads();
  for (int i = tid; i < 16*32; i += 256){
    int r = i >> 5, g = i & 31, c0 = g*4;
    float nd[4];
    gn_norm4(sAct[r][c0+0]+P.l2b[c0+0], sAct[r][c0+1]+P.l2b[c0+1],
             sAct[r][c0+2]+P.l2b[c0+2], sAct[r][c0+3]+P.l2b[c0+3], nd);
#pragma unroll
    for (int j = 0; j < 4; ++j){
      int c = c0 + j;
      sN2[r][c] = nd[j]*P.n2w[c] + P.n2b[c];
    }
  }
  __syncthreads();

  store_c_frag(&sAct[0][0], HD, col0, acct, lane);
  __syncthreads();
  for (int i = tid; i < 16*32; i += 256){
    int r = i >> 5, g = i & 31, c0 = g*4;
    int m = (int)((row0 + r) % MODES);
    const float* mb = P.tmb + m*HD;
    float nd[4];
    gn_norm4(sAct[r][c0+0]+P.tb[c0+0]+mb[c0+0], sAct[r][c0+1]+P.tb[c0+1]+mb[c0+1],
             sAct[r][c0+2]+P.tb[c0+2]+mb[c0+2], sAct[r][c0+3]+P.tb[c0+3]+mb[c0+3], nd);
#pragma unroll
    for (int j = 0; j < 4; ++j){
      int c = c0 + j;
      float nt = nd[j]*P.tnw[c] + P.tnb[c];
      P.out[(row0 + r)*HD + c] = (bf16)eluf(sN2[r][c] + nt);
    }
  }
}

/* ============== output head: GEMM(128->60) + ctrs + select + rotate + transpose ========= */
struct OutP {
  const bf16 *X, *oW;
  const float *ob, *ctrs, *rot, *orig, *den, *fd;
  float *out;
};

__global__ __launch_bounds__(128) void k_out(OutP P){
  __shared__ __align__(16) bf16  sX[16][HD];
  __shared__ __align__(16) float sD[16][64];
  int tid = threadIdx.x, lane = tid & 31, w = tid >> 5;
  long row0 = (long)blockIdx.x * 16;

  for (int i = tid; i < 16*HD; i += 128){
    int r = i >> 7, k = i & 127;
    sX[r][k] = P.X[(row0 + r)*HD + k];
  }
  __syncthreads();

  int col0 = w * 16;
  v8f acc = (v8f)0.0f;
  for (int k0 = 0; k0 < HD; k0 += 32){
    v16bf a = load_a_frag(&sX[0][0], HD, k0, lane);
    v16bf b = load_b_frag(P.oW, HD, col0, k0, lane);
    acc = wmma_bf16(a, b, acc);
  }
  {
    int col = col0 + (lane & 15);
    int rb  = (lane & 16) ? 8 : 0;
    float bv = (col < 2*TSTEP) ? P.ob[col] : 0.f;
#pragma unroll
    for (int r = 0; r < 8; ++r) sD[rb + r][col] = acc[r] + bv;
  }
  __syncthreads();

  for (int i = tid; i < 16*TSTEP; i += 128){
    int r = i / TSTEP, t = i % TSTEP;
    long R = row0 + r;
    int n = (int)(R / MODES), m = (int)(R % MODES);
    bool has_in = P.den[(long)n*KM] > 0.f;
    float fx, fy;
    if (has_in){
      fx = sD[r][2*t + 0] + P.ctrs[n*2 + 0];
      fy = sD[r][2*t + 1] + P.ctrs[n*2 + 1];
    } else {
      long base = (((long)n*MODES + m)*TSTEP + t)*2;
      fx = P.fd[base + 0];
      fy = P.fd[base + 1];
    }
    float px = fx*P.rot[n*4 + 0] + fy*P.rot[n*4 + 2] + P.orig[n*2 + 0];
    float py = fx*P.rot[n*4 + 1] + fy*P.rot[n*4 + 3] + P.orig[n*2 + 1];
    long ob = (((long)n*TSTEP + t)*MODES + m)*2;
    P.out[ob + 0] = px;
    P.out[ob + 1] = py;
  }
}

/* =========================== host launch =========================== */
extern "C" void kernel_launch(void* const* d_in, const int* in_sizes, int n_in,
                              void* d_out, int out_size, void* d_ws, size_t ws_size,
                              hipStream_t stream){
  (void)in_sizes; (void)n_in; (void)out_size; (void)ws_size;

  const float* vN   = (const float*)d_in[IN_VN];
  const float* fd   = (const float*)d_in[IN_FD];
  const float* at   = (const float*)d_in[IN_AT];
  const float* ctrs = (const float*)d_in[IN_CTRS];
  const float* rot  = (const float*)d_in[IN_ROT];
  const float* orig = (const float*)d_in[IN_ORIG];
  const int*   src  = (const int*)d_in[IN_SRC];
  const int*   dst  = (const int*)d_in[IN_DST];

  /* ---- workspace carve (lifetime-aliased big region R) ---- */
  char* w = (char*)d_ws; size_t off = 0;
  auto alloc = [&](size_t b) -> void* { void* p = w + off; off = (off + b + 255) & ~(size_t)255; return p; };

  float* u1    = (float*)alloc(NHEAD*HD*4);
  float* u2    = (float*)alloc(NHEAD*HD*4);
  bf16* l1Wb   = (bf16*)alloc(64*HD*2);
  bf16* tWb    = (bf16*)alloc(64*HD*2);
  bf16* l2Wb   = (bf16*)alloc(HD*HD*2);
  bf16* fl1Wb  = (bf16*)alloc(HD*HD*2);
  bf16* ftWb   = (bf16*)alloc(HD*HD*2);
  bf16* fl2Wb  = (bf16*)alloc(HD*HD*2);
  bf16* oWb    = (bf16*)alloc(HD*64*2);
  bf16* WihB   = (bf16*)alloc(HD*3*HD*2);
  bf16* WhhB   = (bf16*)alloc(HD*3*HD*2);
  bf16* Bfc    = (bf16*)alloc(2*HD*HD*2);
  float* l1mb  = (float*)alloc(MODES*HD*4);
  float* tmb   = (float*)alloc(MODES*HD*4);
  float* den   = (float*)alloc((size_t)N_NODES*KM*4);
  bf16* vnB    = (bf16*)alloc((size_t)N_NODES*MODES*HD*2);
  bf16* aggB   = (bf16*)alloc((size_t)N_NODES*MODES*HD*2);
  bf16* vnoutB = (bf16*)alloc((size_t)N_NODES*MODES*HD*2);
  bf16* hB     = (bf16*)alloc((size_t)N_NODES*MODES*HD*2);

  const size_t SZ_FEATS = (size_t)N_EDGES*MODES*HD*2;          /* 98.3 MB */
  const size_t SZ_EE    = (size_t)N_EDGES*KM*4;                /* 3.07 MB */
  const size_t SZ_MX    = (size_t)N_NODES*KM*4;                /* 0.77 MB */
  const size_t SZ_S     = (size_t)N_NODES*KM*HD*4;             /* 98.3 MB */
  const size_t SZ_GI    = (size_t)N_NODES*MODES*3*HD*4;        /* 147.5 MB */
  const size_t R_SIZE   = 2*SZ_GI;                             /* gi+gh span dominates */
  char* R = (char*)alloc(R_SIZE);
  bf16*     feats = (bf16*)(R);
  float*    ee    = (float*)(R + SZ_FEATS);
  unsigned* mxkey = (unsigned*)(R + SZ_FEATS + SZ_EE);
  float*    sacc  = (float*)(R + SZ_FEATS + SZ_EE + SZ_MX);
  bf16*     sB    = (bf16*)(R + SZ_FEATS + SZ_EE + SZ_MX + SZ_S);
  float*    gi    = (float*)(R);            /* written after feats/ee/mxkey/s dead */
  float*    gh    = (float*)(R + SZ_GI);    /* written after sB dead               */

  /* ---- weight prep (fragment-major bf16 panels) ---- */
  k_prep_u<<<1, 256, 0, stream>>>((const float*)d_in[IN_FC1], (const float*)d_in[IN_FC2],
                                  (const float*)d_in[IN_FC3], u1, u2);
  k_prep_bmat<<<(64*HD+255)/256,   256, 0, stream>>>((const float*)d_in[IN_E_L1W], l1Wb, 128,  60,  60, 128,  64);
  k_prep_bmat<<<(64*HD+255)/256,   256, 0, stream>>>((const float*)d_in[IN_E_TW],  tWb,  128,  60,  60, 128,  64);
  k_prep_bmat<<<(HD*HD+255)/256,   256, 0, stream>>>((const float*)d_in[IN_E_L2W], l2Wb, 128, 128, 128, 128, 128);
  k_prep_bmat<<<(HD*HD+255)/256,   256, 0, stream>>>((const float*)d_in[IN_F_L1W], fl1Wb,128, 134, 128, 128, 128);
  k_prep_bmat<<<(HD*HD+255)/256,   256, 0, stream>>>((const float*)d_in[IN_F_TW],  ftWb, 128, 134, 128, 128, 128);
  k_prep_bmat<<<(HD*HD+255)/256,   256, 0, stream>>>((const float*)d_in[IN_F_L2W], fl2Wb,128, 128, 128, 128, 128);
  k_prep_bmat<<<(HD*64+255)/256,   256, 0, stream>>>((const float*)d_in[IN_F_OW],  oWb,   60, 128, 128,  64, 128);
  k_prep_bmat<<<(HD*3*HD+255)/256, 256, 0, stream>>>((const float*)d_in[IN_WIH],   WihB, 384, 128, 128, 384, 128);
  k_prep_bmat<<<(HD*3*HD+255)/256, 256, 0, stream>>>((const float*)d_in[IN_WHH],   WhhB, 384, 128, 128, 384, 128);
  k_prep_bfc<<<(2*HD*HD+255)/256,  256, 0, stream>>>((const float*)d_in[IN_FC1], Bfc);
  k_prep_modebias<<<(MODES*HD+255)/256, 256, 0, stream>>>((const float*)d_in[IN_F_L1W], l1mb);
  k_prep_modebias<<<(MODES*HD+255)/256, 256, 0, stream>>>((const float*)d_in[IN_F_TW],  tmb);
  {
    long n = (long)N_NODES*MODES*HD;
    k_f32_to_bf16<<<(unsigned)((n + 255)/256), 256, 0, stream>>>(vN, vnB, n);
  }

  hipMemsetAsync(mxkey, 0, SZ_MX, stream);
  hipMemsetAsync(den,   0, (size_t)N_NODES*KM*4, stream);
  hipMemsetAsync(sacc,  0, SZ_S, stream);

  /* ---- edge encoder ---- */
  EncP ep;
  ep.fd = fd; ep.ctrs = ctrs; ep.at = at;
  ep.atW = (const float*)d_in[IN_ATW]; ep.atB = (const float*)d_in[IN_ATB];
  ep.src = src; ep.dst = dst;
  ep.l1W = l1Wb; ep.l2W = l2Wb; ep.tW = tWb;
  ep.l1b = (const float*)d_in[IN_E_L1B]; ep.n1w = (const float*)d_in[IN_E_N1W]; ep.n1b = (const float*)d_in[IN_E_N1B];
  ep.l2b = (const float*)d_in[IN_E_L2B]; ep.n2w = (const float*)d_in[IN_E_N2W]; ep.n2b = (const float*)d_in[IN_E_N2B];
  ep.tb  = (const float*)d_in[IN_E_TB];  ep.tnw = (const float*)d_in[IN_E_TNW]; ep.tnb = (const float*)d_in[IN_E_TNB];
  ep.feats = feats;
  k_edge_enc<<<N_EDGES*MODES/16, 256, 0, stream>>>(ep);

  /* ---- attention ---- */
  k_escore<<<(N_EDGES*MODES + 255)/256, 256, 0, stream>>>(feats, vN, dst, u1, u2, ee, mxkey);
  k_expsum<<<(N_EDGES*KM + 255)/256, 256, 0, stream>>>(dst, mxkey, ee, den);
  k_scatter<<<N_EDGES*MODES, 128, 0, stream>>>(feats, ee, den, dst, sacc);
  {
    long n = (long)N_NODES*MODES*2*HD;
    k_pack_s<<<(unsigned)((n + 255)/256), 256, 0, stream>>>(sacc, sB);
  }

  /* ---- agg = (fc1 applied after segment sum), then GRU GEMMs ---- */
  k_gemm_bf16<<<dim3(N_NODES*MODES/16, 1), 256, 16*256*2, stream>>>(sB, 256, Bfc, nullptr, nullptr, aggB, 128);
  k_gemm_bf16<<<dim3(N_NODES*MODES/16, 3), 256, 16*128*2, stream>>>(aggB, 128, WihB,
                                              (const float*)d_in[IN_BIH], gi, nullptr, 384);
  k_gemm_bf16<<<dim3(N_NODES*MODES/16, 3), 256, 16*128*2, stream>>>(vnB, 128, WhhB,
                                              (const float*)d_in[IN_BHH], gh, nullptr, 384);
  {
    long n = (long)N_NODES*MODES*HD;
    k_gru<<<(unsigned)((n + 255)/256), 256, 0, stream>>>(gi, gh, vN, den, vnoutB);
  }

  /* ---- fout LinearRes ---- */
  FoutP fp;
  fp.X = vnoutB; fp.l1W = fl1Wb; fp.l2W = fl2Wb; fp.tW = ftWb;
  fp.l1b = (const float*)d_in[IN_F_L1B]; fp.l1mb = l1mb;
  fp.n1w = (const float*)d_in[IN_F_N1W]; fp.n1b = (const float*)d_in[IN_F_N1B];
  fp.l2b = (const float*)d_in[IN_F_L2B]; fp.n2w = (const float*)d_in[IN_F_N2W]; fp.n2b = (const float*)d_in[IN_F_N2B];
  fp.tb  = (const float*)d_in[IN_F_TB];  fp.tmb = tmb;
  fp.tnw = (const float*)d_in[IN_F_TNW]; fp.tnb = (const float*)d_in[IN_F_TNB];
  fp.out = hB;
  k_fout_enc<<<N_NODES*MODES/16, 256, 0, stream>>>(fp);

  /* ---- output head ---- */
  OutP op;
  op.X = hB; op.oW = oWb; op.ob = (const float*)d_in[IN_F_OB];
  op.ctrs = ctrs; op.rot = rot; op.orig = orig; op.den = den; op.fd = fd;
  op.out = (float*)d_out;
  k_out<<<N_NODES*MODES/16, 128, 0, stream>>>(op);
}
